// JiRackPyTorch_42460046688342
// MI455X (gfx1250) — matrix operand order repaired and checked by reading, not compiled, not run on hardware
//
#include <hip/hip_runtime.h>
#include <stdint.h>
#include <stddef.h>

// ---------------- problem constants (match reference) ----------------
constexpr int D_   = 2048;    // model dim
constexpr int L_   = 4;       // layers
constexpr int H_   = 32;      // heads
constexpr int HD_  = 64;      // head dim
constexpr int FF_  = 8192;    // ffn hidden
constexpr int V_   = 50257;   // vocab
constexpr int B_   = 4;
constexpr int T_   = 512;
constexpr int MTOK = B_ * T_;     // 2048 tokens
constexpr int VPAD = 50304;       // V rounded to 128: pad rows so lm-head B staging never faults

typedef __attribute__((ext_vector_type(16))) __bf16 v16bf;
typedef __attribute__((ext_vector_type(8)))  float  v8f;
typedef int gv4i __attribute__((vector_size(16)));   // matches async builtin pointee type

#if defined(__AMDGCN__) && __has_builtin(__builtin_amdgcn_global_load_async_to_lds_b128)
#define HAVE_ASYNC_LDS 1
#else
#define HAVE_ASYNC_LDS 0
#endif

#define ASYNC_G(p) ((__attribute__((address_space(1))) gv4i*)(p))
#define ASYNC_L(p) ((__attribute__((address_space(3))) gv4i*)(p))

__device__ __forceinline__ void wait_async_all() {
#if HAVE_ASYNC_LDS
  __builtin_amdgcn_s_wait_asynccnt(0);
#endif
}

__device__ __forceinline__ void sched_fence() {
#if defined(__AMDGCN__) && __has_builtin(__builtin_amdgcn_sched_barrier)
  __builtin_amdgcn_sched_barrier(0);   // keep next iteration's waits below the WMMA burst
#endif
}

__device__ __forceinline__ unsigned short f2bf(float f) {
  unsigned int u = __float_as_uint(f);
  u += 0x7FFFu + ((u >> 16) & 1u);           // round-to-nearest-even
  return (unsigned short)(u >> 16);
}

// =====================================================================
// bf16 WMMA GEMM:  C[M,N](f32) (+)= alpha * A[M,K] @ Bt[N,K]^T   (both bf16)
//   A row-major (lda==K), Bt row-major K-major (ldb==K), C row-major (ldc==N).
//   Batched via gridDim.z (element strides sA/sB/sC).
//   EPI: 0 store, 1 accumulate (residual add), 2 scale+causal mask.
// K-major B means A and B tiles stage identically: 128-bit async copies
// straight into LDS (global_load_async_to_lds_b128, ASYNCcnt), double
// buffered; fragments are two conflict-free ds_load_b128 per operand.
// Pipeline per K-step: wait prev asyncs -> barrier -> issue next asyncs
// -> fragment ds_loads -> 8x WMMA (async copies in flight underneath).
// =====================================================================
enum { EPI_STORE = 0, EPI_ADD = 1, EPI_MASK = 2 };

union Frag32 { uint4 u[2]; v16bf v; };

template<int BM, int BN, int WM, int WN, int EPI>
__global__ __launch_bounds__(256)
void gemm_bf16_kernel(const unsigned short* __restrict__ A,
                      const unsigned short* __restrict__ Bt,
                      float* __restrict__ C,
                      int M, int N, int K,
                      long long sA, long long sB, long long sC,
                      float alpha)
{
  static_assert(BM == 128, "staging assumes BM==128 with 256 threads");
  constexpr int BK = 32;
  constexpr int WAVES_N = BN / WN;
  constexpr int MI = WM / 16, NI = WN / 16;
  constexpr int SA = 40;   // halves per LDS row (32 + pad; 80B rows keep b128 reads conflict-free)

  __shared__ unsigned short lAs[2][BM * SA];
  __shared__ unsigned short lBs[2][BN * SA];

  const int z = blockIdx.z;
  A  += (long long)z * sA;
  Bt += (long long)z * sB;
  C  += (long long)z * sC;

  const int bm   = blockIdx.y * BM;
  const int bn   = blockIdx.x * BN;
  const int tid  = threadIdx.x;
  const int wave = tid >> 5;
  const int lane = tid & 31;
  const int g    = lane >> 4;     // half-wave selects K sub-range per ISA layout
  const int r    = lane & 15;
  const int wm   = (wave / WAVES_N) * WM;
  const int wn   = (wave % WAVES_N) * WN;

  // staging: each thread owns one 16-half row segment of the A tile and
  // (if tid < 2*BN) one of the B tile.
  const int arow = tid >> 1, aseg = (tid & 1) * 16;
  const bool bAct = (tid < 2 * BN);

#if !HAVE_ASYNC_LDS
  uint4 sa0, sa1, sb0, sb1;
#endif

  auto stage_issue = [&](int kb, int buf) {
    const unsigned short* gpA = A + (long long)(bm + arow) * K + kb + aseg;
#if HAVE_ASYNC_LDS
    unsigned short* lpA = &lAs[buf][arow * SA + aseg];
    __builtin_amdgcn_global_load_async_to_lds_b128(ASYNC_G(gpA),     ASYNC_L(lpA),     0, 0);
    __builtin_amdgcn_global_load_async_to_lds_b128(ASYNC_G(gpA + 8), ASYNC_L(lpA + 8), 0, 0);
    if (bAct) {
      const unsigned short* gpB = Bt + (long long)(bn + arow) * K + kb + aseg;
      unsigned short* lpB = &lBs[buf][arow * SA + aseg];
      __builtin_amdgcn_global_load_async_to_lds_b128(ASYNC_G(gpB),     ASYNC_L(lpB),     0, 0);
      __builtin_amdgcn_global_load_async_to_lds_b128(ASYNC_G(gpB + 8), ASYNC_L(lpB + 8), 0, 0);
    }
#else
    sa0 = ((const uint4*)gpA)[0];
    sa1 = ((const uint4*)gpA)[1];
    if (bAct) {
      const unsigned short* gpB = Bt + (long long)(bn + arow) * K + kb + aseg;
      sb0 = ((const uint4*)gpB)[0];
      sb1 = ((const uint4*)gpB)[1];
    }
#endif
    (void)buf;
  };
#if !HAVE_ASYNC_LDS
  auto stage_commit = [&](int buf) {
    uint4* lpA = (uint4*)&lAs[buf][arow * SA + aseg];
    lpA[0] = sa0; lpA[1] = sa1;
    if (bAct) {
      uint4* lpB = (uint4*)&lBs[buf][arow * SA + aseg];
      lpB[0] = sb0; lpB[1] = sb1;
    }
  };
#endif

  v8f acc[MI][NI] = {};

  stage_issue(0, 0);
#if !HAVE_ASYNC_LDS
  stage_commit(0);
#endif

  for (int kb = 0; kb < K; kb += BK) {
    const int cur = (kb >> 5) & 1;
    wait_async_all();            // previous async stage landed in LDS
    __syncthreads();

    const bool more = (kb + BK) < K;
    if (more) stage_issue(kb + BK, cur ^ 1);      // DMA for next tile flies under this step

    // fragment loads (ISA 16-bit wave32 layouts)
    Frag32 fa[MI], fb[NI];
#pragma unroll
    for (int mi = 0; mi < MI; mi++) {
      // A operand: lane(g,r) holds K = 8g..8g+7 and 16+8g..16+8g+7 of row r
      const unsigned short* base = &lAs[cur][(wm + mi * 16 + r) * SA + g * 8];
      fa[mi].u[0] = *(const uint4*)base;
      fa[mi].u[1] = *(const uint4*)(base + 16);
    }
#pragma unroll
    for (int ni = 0; ni < NI; ni++) {
      // B operand: lane(g,r) holds K = 8g..8g+7 and 16+8g..16+8g+7 of column r (K-major row)
      const unsigned short* base = &lBs[cur][(wn + ni * 16 + r) * SA + g * 8];
      fb[ni].u[0] = *(const uint4*)base;
      fb[ni].u[1] = *(const uint4*)(base + 16);
    }

#pragma unroll
    for (int mi = 0; mi < MI; mi++)
#pragma unroll
      for (int ni = 0; ni < NI; ni++)
        acc[mi][ni] = __builtin_amdgcn_wmma_f32_16x16x32_bf16(
            false, fa[mi].v, false, fb[ni].v, (short)0, acc[mi][ni], false, false);

#if !HAVE_ASYNC_LDS
    if (more) stage_commit(cur ^ 1);
#endif
    sched_fence();   // forbid hoisting next iteration's s_wait_asynccnt above the WMMAs
  }

  // ---- epilogue: C layout = VGPR v -> row v + 8g, col = lane&15
#pragma unroll
  for (int mi = 0; mi < MI; mi++)
#pragma unroll
    for (int ni = 0; ni < NI; ni++) {
      int colg = bn + wn + ni * 16 + r;
      if (colg < N) {
#pragma unroll
        for (int vr = 0; vr < 8; vr++) {
          int rowg = bm + wm + mi * 16 + vr + 8 * g;
          float val = acc[mi][ni][vr] * alpha;
          if (EPI == EPI_MASK && colg > rowg) val = -1.0e30f;   // causal mask
          long long off = (long long)rowg * N + colg;
          if (EPI == EPI_ADD) C[off] += val;
          else                C[off]  = val;
        }
      }
    }
}

// =====================================================================
// Support kernels
// =====================================================================
__global__ __launch_bounds__(256)
void convert_bf16_kernel(const float* __restrict__ in, unsigned short* __restrict__ out, long long n) {
  long long i = (long long)blockIdx.x * 256 + threadIdx.x;
  if (i < n) out[i] = f2bf(in[i]);
}

// src [z][R][C] f32 -> dst [z][C][R] bf16 (32x32 LDS tiles; R,C multiples of 32)
__global__ __launch_bounds__(256)
void transpose_conv_kernel(const float* __restrict__ src, unsigned short* __restrict__ dst,
                           int R, int C) {
  __shared__ float tile[32][33];
  long long zoff = (long long)blockIdx.z * R * C;
  int c0 = blockIdx.x * 32;
  int r0 = blockIdx.y * 32;
  int tx = threadIdx.x & 31, ty = threadIdx.x >> 5;        // 32 x 8
  for (int yy = ty; yy < 32; yy += 8)
    tile[yy][tx] = src[zoff + (long long)(r0 + yy) * C + c0 + tx];
  __syncthreads();
  for (int yy = ty; yy < 32; yy += 8)
    dst[zoff + (long long)(c0 + yy) * R + r0 + tx] = f2bf(tile[tx][yy]);
}

__global__ __launch_bounds__(256)
void embed_kernel(const int* __restrict__ ids, const float* __restrict__ emb, float* __restrict__ x) {
  int m = blockIdx.x;
  long long row = (long long)ids[m] * D_;
  for (int c = threadIdx.x; c < D_; c += 256) x[(long long)m * D_ + c] = emb[row + c];
}

__global__ __launch_bounds__(256)
void rmsnorm_kernel(const float* __restrict__ x, const float* __restrict__ w,
                    unsigned short* __restrict__ out) {
  __shared__ float red[256];
  long long row = (long long)blockIdx.x * D_;
  float s = 0.f;
  for (int c = threadIdx.x; c < D_; c += 256) { float v = x[row + c]; s += v * v; }
  red[threadIdx.x] = s;
  __syncthreads();
  for (int off = 128; off > 0; off >>= 1) {
    if (threadIdx.x < off) red[threadIdx.x] += red[threadIdx.x + off];
    __syncthreads();
  }
  float inv = rsqrtf(red[0] * (1.0f / D_) + 1e-6f);
  for (int c = threadIdx.x; c < D_; c += 256) out[row + c] = f2bf(x[row + c] * inv * w[c]);
}

// RoPE q,k + reorder: q,k -> [bh,t,hd] (k is the K-major B of QK^T),
// v -> [bh,hd,t] (K-major B of P@V).
__global__ __launch_bounds__(256)
void rope_reorder_kernel(const float* __restrict__ qf, const float* __restrict__ kf,
                         const float* __restrict__ vf,
                         unsigned short* __restrict__ qbo, unsigned short* __restrict__ kbo,
                         unsigned short* __restrict__ vbt) {
  int m = blockIdx.x;                 // token 0..B*T-1
  int b = m / T_, t = m % T_;
  for (int p = threadIdx.x; p < D_ / 2; p += 256) {
    int h = p >> 5;                   // 32 pairs per head
    int i = p & 31;
    int d = 2 * i;
    float freq = __powf(10000.0f, -(float)d * (1.0f / HD_));
    float sn, cs;
    __sincosf((float)t * freq, &sn, &cs);
    long long src = (long long)m * D_ + h * HD_ + d;
    float qa = qf[src], qb = qf[src + 1];
    float ka = kf[src], kb = kf[src + 1];
    long long bh = (long long)b * H_ + h;
    long long qd = (bh * T_ + t) * HD_ + d;
    qbo[qd]     = f2bf(qa * cs - qb * sn);
    qbo[qd + 1] = f2bf(qa * sn + qb * cs);
    kbo[qd]     = f2bf(ka * cs - kb * sn);
    kbo[qd + 1] = f2bf(ka * sn + kb * cs);
  }
  for (int e = threadIdx.x; e < D_; e += 256) {
    int h = e >> 6, d = e & 63;
    vbt[(((long long)b * H_ + h) * HD_ + d) * T_ + t] = f2bf(vf[(long long)m * D_ + e]);
  }
}

// row softmax over T_ cols: f32 scores -> bf16 probabilities
__global__ __launch_bounds__(256)
void softmax_kernel(const float* __restrict__ S, unsigned short* __restrict__ P) {
  __shared__ float red[256];
  long long row = (long long)blockIdx.x * T_;
  float m = -3.4e38f;
  for (int c = threadIdx.x; c < T_; c += 256) m = fmaxf(m, S[row + c]);
  red[threadIdx.x] = m;
  __syncthreads();
  for (int off = 128; off > 0; off >>= 1) {
    if (threadIdx.x < off) red[threadIdx.x] = fmaxf(red[threadIdx.x], red[threadIdx.x + off]);
    __syncthreads();
  }
  m = red[0];
  __syncthreads();
  float s = 0.f;
  for (int c = threadIdx.x; c < T_; c += 256) s += __expf(S[row + c] - m);
  red[threadIdx.x] = s;
  __syncthreads();
  for (int off = 128; off > 0; off >>= 1) {
    if (threadIdx.x < off) red[threadIdx.x] += red[threadIdx.x + off];
    __syncthreads();
  }
  float inv = 1.0f / red[0];
  for (int c = threadIdx.x; c < T_; c += 256)
    P[row + c] = f2bf(__expf(S[row + c] - m) * inv);
}

// of [bh,t,hd] f32 -> ob [token, D] bf16
__global__ __launch_bounds__(256)
void reorder_o_kernel(const float* __restrict__ of, unsigned short* __restrict__ ob) {
  long long idx = (long long)blockIdx.x * 256 + threadIdx.x;   // B*H*T*HD total
  int d = idx & 63;
  long long rr = idx >> 6;
  int t = (int)(rr % T_);
  long long bh = rr / T_;
  int b = (int)(bh / H_), h = (int)(bh % H_);
  ob[((long long)b * T_ + t) * D_ + h * HD_ + d] = f2bf(of[idx]);
}

__global__ __launch_bounds__(256)
void silu_mul_kernel(const float* __restrict__ g1, const float* __restrict__ g3,
                     unsigned short* __restrict__ gb, long long n) {
  long long i = (long long)blockIdx.x * 256 + threadIdx.x;
  if (i < n) {
    float a = g1[i];
    float s = a / (1.0f + __expf(-a));
    gb[i] = f2bf(s * g3[i]);
  }
}

// =====================================================================
// Host orchestration
// =====================================================================
extern "C" void kernel_launch(void* const* d_in, const int* in_sizes, int n_in,
                              void* d_out, int out_size, void* d_ws, size_t ws_size,
                              hipStream_t stream) {
  (void)in_sizes; (void)n_in; (void)out_size; (void)ws_size;
  const int*   ids  = (const int*)  d_in[0];
  const float* emb  = (const float*)d_in[1];
  const float* wq   = (const float*)d_in[2];
  const float* wk   = (const float*)d_in[3];
  const float* wv   = (const float*)d_in[4];
  const float* wo   = (const float*)d_in[5];
  const float* w1   = (const float*)d_in[6];
  const float* w3   = (const float*)d_in[7];
  const float* w2   = (const float*)d_in[8];
  const float* n1   = (const float*)d_in[9];
  const float* n2   = (const float*)d_in[10];
  const float* lnf  = (const float*)d_in[11];
  float* logits = (float*)d_out;

  // ---- workspace arena ----
  char* p = (char*)d_ws;
  auto alloc = [&](size_t bytes) -> void* {
    void* r = (void*)p;
    p += (bytes + 255) & ~(size_t)255;
    return r;
  };
  const size_t DD = (size_t)D_ * D_;
  // all weights stored transposed ([N,K] K-major) in bf16
  unsigned short* wqB  = (unsigned short*)alloc((size_t)L_ * DD * 2);
  unsigned short* wkB  = (unsigned short*)alloc((size_t)L_ * DD * 2);
  unsigned short* wvB  = (unsigned short*)alloc((size_t)L_ * DD * 2);
  unsigned short* woB  = (unsigned short*)alloc((size_t)L_ * DD * 2);
  unsigned short* w1B  = (unsigned short*)alloc((size_t)L_ * D_ * FF_ * 2);
  unsigned short* w3B  = (unsigned short*)alloc((size_t)L_ * D_ * FF_ * 2);
  unsigned short* w2B  = (unsigned short*)alloc((size_t)L_ * FF_ * D_ * 2);
  unsigned short* embB = (unsigned short*)alloc((size_t)VPAD * D_ * 2);  // [V,D] is already K-major for tied head
  float*          x    = (float*)alloc((size_t)MTOK * D_ * 4);
  unsigned short* hb   = (unsigned short*)alloc((size_t)MTOK * D_ * 2);
  float*          qf   = (float*)alloc((size_t)MTOK * D_ * 4);
  float*          kf   = (float*)alloc((size_t)MTOK * D_ * 4);
  float*          vf   = (float*)alloc((size_t)MTOK * D_ * 4);
  unsigned short* qb   = (unsigned short*)alloc((size_t)MTOK * D_ * 2);
  unsigned short* kb   = (unsigned short*)alloc((size_t)MTOK * D_ * 2);
  unsigned short* vbt  = (unsigned short*)alloc((size_t)MTOK * D_ * 2);
  float*          S    = (float*)alloc((size_t)B_ * H_ * T_ * T_ * 4);
  unsigned short* P    = (unsigned short*)alloc((size_t)B_ * H_ * T_ * T_ * 2);
  float*          of   = (float*)alloc((size_t)MTOK * D_ * 4);
  unsigned short* ob   = (unsigned short*)alloc((size_t)MTOK * D_ * 2);
  float*          g1   = (float*)alloc((size_t)MTOK * FF_ * 4);
  float*          g3   = (float*)alloc((size_t)MTOK * FF_ * 4);
  unsigned short* gb   = (unsigned short*)alloc((size_t)MTOK * FF_ * 2);

  // ---- per-call weight prep: f32 [K,N] -> bf16 [N,K] ----
  transpose_conv_kernel<<<dim3(D_ / 32, D_ / 32, L_), 256, 0, stream>>>(wq, wqB, D_, D_);
  transpose_conv_kernel<<<dim3(D_ / 32, D_ / 32, L_), 256, 0, stream>>>(wk, wkB, D_, D_);
  transpose_conv_kernel<<<dim3(D_ / 32, D_ / 32, L_), 256, 0, stream>>>(wv, wvB, D_, D_);
  transpose_conv_kernel<<<dim3(D_ / 32, D_ / 32, L_), 256, 0, stream>>>(wo, woB, D_, D_);
  transpose_conv_kernel<<<dim3(FF_ / 32, D_ / 32, L_), 256, 0, stream>>>(w1, w1B, D_, FF_);
  transpose_conv_kernel<<<dim3(FF_ / 32, D_ / 32, L_), 256, 0, stream>>>(w3, w3B, D_, FF_);
  transpose_conv_kernel<<<dim3(D_ / 32, FF_ / 32, L_), 256, 0, stream>>>(w2, w2B, FF_, D_);
  convert_bf16_kernel<<<(unsigned)(((long long)V_ * D_ + 255) / 256), 256, 0, stream>>>(
      emb, embB, (long long)V_ * D_);

  // ---- embedding gather ----
  embed_kernel<<<MTOK, 256, 0, stream>>>(ids, emb, x);

  const float scale = 0.125f;   // HD^-0.5
  for (int l = 0; l < L_; l++) {
    const unsigned short* wql = wqB + (size_t)l * DD;
    const unsigned short* wkl = wkB + (size_t)l * DD;
    const unsigned short* wvl = wvB + (size_t)l * DD;
    const unsigned short* wol = woB + (size_t)l * DD;
    const unsigned short* w1l = w1B + (size_t)l * D_ * FF_;
    const unsigned short* w3l = w3B + (size_t)l * D_ * FF_;
    const unsigned short* w2l = w2B + (size_t)l * FF_ * D_;

    rmsnorm_kernel<<<MTOK, 256, 0, stream>>>(x, n1 + (size_t)l * D_, hb);

    dim3 gP(D_ / 128, MTOK / 128, 1);
    gemm_bf16_kernel<128,128,32,64,EPI_STORE><<<gP, 256, 0, stream>>>(
        hb, wql, qf, MTOK, D_, D_, 0, 0, 0, 1.0f);
    gemm_bf16_kernel<128,128,32,64,EPI_STORE><<<gP, 256, 0, stream>>>(
        hb, wkl, kf, MTOK, D_, D_, 0, 0, 0, 1.0f);
    gemm_bf16_kernel<128,128,32,64,EPI_STORE><<<gP, 256, 0, stream>>>(
        hb, wvl, vf, MTOK, D_, D_, 0, 0, 0, 1.0f);

    rope_reorder_kernel<<<MTOK, 256, 0, stream>>>(qf, kf, vf, qb, kb, vbt);

    // scores = Q K^T * scale + causal mask (batched over B*H); B = K [t,hd] K-major
    dim3 gS(T_ / 128, T_ / 128, B_ * H_);
    gemm_bf16_kernel<128,128,32,64,EPI_MASK><<<gS, 256, 0, stream>>>(
        qb, kb, S, T_, T_, HD_,
        (long long)T_ * HD_, (long long)T_ * HD_, (long long)T_ * T_, scale);

    softmax_kernel<<<B_ * H_ * T_, 256, 0, stream>>>(S, P);

    // O = P @ V (batched, N=64); B = V^T [hd,t] K-major
    dim3 gO(1, T_ / 128, B_ * H_);
    gemm_bf16_kernel<128,64,32,32,EPI_STORE><<<gO, 256, 0, stream>>>(
        P, vbt, of, T_, HD_, T_,
        (long long)T_ * T_, (long long)HD_ * T_, (long long)T_ * HD_, 1.0f);

    reorder_o_kernel<<<(MTOK * D_) / 256, 256, 0, stream>>>(of, ob);

    // x += O @ Wo
    gemm_bf16_kernel<128,128,32,64,EPI_ADD><<<gP, 256, 0, stream>>>(
        ob, wol, x, MTOK, D_, D_, 0, 0, 0, 1.0f);

    // FFN
    rmsnorm_kernel<<<MTOK, 256, 0, stream>>>(x, n2 + (size_t)l * D_, hb);
    dim3 gF(FF_ / 128, MTOK / 128, 1);
    gemm_bf16_kernel<128,128,32,64,EPI_STORE><<<gF, 256, 0, stream>>>(
        hb, w1l, g1, MTOK, FF_, D_, 0, 0, 0, 1.0f);
    gemm_bf16_kernel<128,128,32,64,EPI_STORE><<<gF, 256, 0, stream>>>(
        hb, w3l, g3, MTOK, FF_, D_, 0, 0, 0, 1.0f);
    silu_mul_kernel<<<(unsigned)(((long long)MTOK * FF_ + 255) / 256), 256, 0, stream>>>(
        g1, g3, gb, (long long)MTOK * FF_);
    gemm_bf16_kernel<128,128,32,64,EPI_ADD><<<gP, 256, 0, stream>>>(
        gb, w2l, x, MTOK, D_, FF_, 0, 0, 0, 1.0f);
  }

  // final norm + tied lm_head: logits = rmsnorm(x) @ emb^T ; B = emb [V,D] K-major
  rmsnorm_kernel<<<MTOK, 256, 0, stream>>>(x, lnf, hb);
  dim3 gL((V_ + 127) / 128, MTOK / 128, 1);
  gemm_bf16_kernel<128,128,32,64,EPI_STORE><<<gL, 256, 0, stream>>>(
      hb, embB, logits, MTOK, V_, D_, 0, 0, 0, 1.0f);
}
